// MultiHeadAttentionLayer_11304353923194
// MI455X (gfx1250) — matrix operand order repaired and checked by reading, compiled, not run on hardware
//
#include <hip/hip_runtime.h>
#include <hip/hip_bf16.h>

typedef __bf16 bf16_t;
typedef __attribute__((ext_vector_type(16))) __bf16 v16bf;
typedef __attribute__((ext_vector_type(8)))  __bf16 v8bf;
typedef __attribute__((ext_vector_type(8)))  float  v8f;

// ---------------------------------------------------------------------------
// WMMA fragment helpers (contiguous k order k = 16h+s in A and B, wave32)
// ---------------------------------------------------------------------------

// A operand: 16(M) x 32(K) bf16, source row-major with leading dim `ld`.
// lane<16: row=lane,    elems 0..15 = K0..15
// lane>=16: row=lane-16, elems 0..15 = K16..31
__device__ __forceinline__ v16bf load_frag_a(const bf16_t* __restrict__ src, int ld) {
  int lane = threadIdx.x & 31;
  int row  = lane & 15;
  int hi   = lane >> 4;
  const bf16_t* p = src + (size_t)row * ld + hi * 16;
  v8bf lo = *(const v8bf*)(p);
  v8bf up = *(const v8bf*)(p + 8);
  v16bf a;
#pragma unroll
  for (int e = 0; e < 8; ++e) { a[e] = lo[e]; a[e + 8] = up[e]; }
  return a;
}

// B operand: 32(K) x 16(N) bf16, source stored N-major / K-contiguous
// (i.e. srcT[n*ld + k] == B[k][n]).
// lane<16: col=lane, elems 0..15 = K0..15 ; lane>=16: col=lane-16, K16..31
__device__ __forceinline__ v16bf load_frag_b(const bf16_t* __restrict__ srcT, int ld) {
  int lane = threadIdx.x & 31;
  int n    = lane & 15;
  int hi   = lane >> 4;
  return *(const v16bf*)(srcT + (size_t)n * ld + hi * 16);
}

__device__ __forceinline__ v8f wmma_bf16(v16bf a, v16bf b, v8f c) {
  return __builtin_amdgcn_wmma_f32_16x16x32_bf16(false, a, false, b, (short)0, c,
                                                 false, false);
}

// ---------------------------------------------------------------------------
// Constants
// ---------------------------------------------------------------------------
#define BDIM 2
#define SDIM 2048
#define EDIM 1024
#define HDIM 16
#define DDIM 64
#define ROWS (BDIM * SDIM)   // 4096

// ---------------------------------------------------------------------------
// 1) LayerNorm: x f32 [4096,1024] -> xn bf16 [4096,1024]
// ---------------------------------------------------------------------------
__global__ void ln_kernel(const float* __restrict__ x,
                          const float* __restrict__ g,
                          const float* __restrict__ bta,
                          bf16_t* __restrict__ xn) {
  int row = blockIdx.x;
  int tid = threadIdx.x;                 // 256 threads, 4 elems each
  const float* xr = x + (size_t)row * EDIM;
  float4 v = *(const float4*)(xr + tid * 4);
  float s  = v.x + v.y + v.z + v.w;
  float s2 = v.x * v.x + v.y * v.y + v.z * v.z + v.w * v.w;
#pragma unroll
  for (int off = 16; off; off >>= 1) {
    s  += __shfl_xor(s, off, 32);
    s2 += __shfl_xor(s2, off, 32);
  }
  __shared__ float red0[8], red1[8];
  int w = tid >> 5, lane = tid & 31;
  if (lane == 0) { red0[w] = s; red1[w] = s2; }
  __syncthreads();
  if (w == 0) {
    float a = (lane < 8) ? red0[lane] : 0.f;
    float b = (lane < 8) ? red1[lane] : 0.f;
#pragma unroll
    for (int off = 4; off; off >>= 1) {
      a += __shfl_xor(a, off, 32);
      b += __shfl_xor(b, off, 32);
    }
    if (lane == 0) { red0[0] = a; red1[0] = b; }
  }
  __syncthreads();
  float mean = red0[0] * (1.f / EDIM);
  float var  = red1[0] * (1.f / EDIM) - mean * mean;
  float rstd = rsqrtf(var + 1e-5f);
  float4 gg = *(const float4*)(g + tid * 4);
  float4 bb = *(const float4*)(bta + tid * 4);
  bf16_t* o = xn + (size_t)row * EDIM + tid * 4;
  o[0] = (bf16_t)((v.x - mean) * rstd * gg.x + bb.x);
  o[1] = (bf16_t)((v.y - mean) * rstd * gg.y + bb.y);
  o[2] = (bf16_t)((v.z - mean) * rstd * gg.z + bb.z);
  o[3] = (bf16_t)((v.w - mean) * rstd * gg.w + bb.w);
}

// ---------------------------------------------------------------------------
// 2) Weight transpose + f32->bf16: W[K=1024][N=1024] -> Wt[N][K] bf16
// ---------------------------------------------------------------------------
__global__ void wtrans_kernel(const float* __restrict__ w0, const float* __restrict__ w1,
                              const float* __restrict__ w2, const float* __restrict__ w3,
                              bf16_t* __restrict__ t0, bf16_t* __restrict__ t1,
                              bf16_t* __restrict__ t2, bf16_t* __restrict__ t3) {
  const float* src; bf16_t* dst;
  switch (blockIdx.z) {
    case 0: src = w0; dst = t0; break;
    case 1: src = w1; dst = t1; break;
    case 2: src = w2; dst = t2; break;
    default: src = w3; dst = t3; break;
  }
  __shared__ float tile[32][33];
  int bx = blockIdx.x * 32;   // K block
  int by = blockIdx.y * 32;   // N block
  int tx = threadIdx.x & 31;
  int ty = threadIdx.x >> 5;  // 8 row-groups
#pragma unroll
  for (int r = ty; r < 32; r += 8)
    tile[r][tx] = src[(size_t)(bx + r) * EDIM + by + tx];
  __syncthreads();
#pragma unroll
  for (int r = ty; r < 32; r += 8)
    dst[(size_t)(by + r) * EDIM + bx + tx] = (bf16_t)tile[tx][r];
}

// ---------------------------------------------------------------------------
// 3) WMMA GEMM: C[4096 x 1024] = A[4096 x 1024] * Bt^T  (Bt is [N][K] bf16)
//    One wave -> 32(M) x 64(N) tile: 2 A frags reused over 4 B tiles.
//    8 waves / block, 256 blocks.
//    MODE 0: +bias, write bf16 to [B,H,S,D]   (Q, K)
//    MODE 1: +bias, write bf16 to [B,H,D,S]   (V, pre-transposed)
//    MODE 2: +bias +residual, write f32 to d_out (output projection)
// ---------------------------------------------------------------------------
template <int MODE>
__global__ void gemm_kernel(const bf16_t* __restrict__ A,
                            const bf16_t* __restrict__ Bt,
                            const float* __restrict__ bias,
                            const float* __restrict__ resid,
                            bf16_t* __restrict__ out_bf,
                            float* __restrict__ out_f) {
  int w      = blockIdx.x * 8 + (threadIdx.x >> 5);
  int mtile2 = w >> 4;   // 0..127 (32 rows each)
  int ngrp   = w & 15;   // 0..15  (64 columns each)
  const bf16_t* arow0 = A + (size_t)(mtile2 * 32) * EDIM;
  const bf16_t* arow1 = arow0 + (size_t)16 * EDIM;

  v8f acc0[4] = {};
  v8f acc1[4] = {};
  for (int k0 = 0; k0 < EDIM; k0 += 32) {
    if (k0 + 32 < EDIM) {
      __builtin_prefetch(arow0 + k0 + 32, 0, 1);   // global_prefetch_b8
      __builtin_prefetch(arow1 + k0 + 32, 0, 1);
    }
    v16bf a0 = load_frag_a(arow0 + k0, EDIM);
    v16bf a1 = load_frag_a(arow1 + k0, EDIM);
#pragma unroll
    for (int t = 0; t < 4; ++t) {
      v16bf b = load_frag_b(Bt + (size_t)(ngrp * 64 + t * 16) * EDIM + k0, EDIM);
      acc0[t] = wmma_bf16(a0, b, acc0[t]);
      acc1[t] = wmma_bf16(a1, b, acc1[t]);
    }
  }

  int lane = threadIdx.x & 31;
  int nlo  = lane & 15;
  int hi   = lane >> 4;
#pragma unroll
  for (int half = 0; half < 2; ++half) {
    v8f* acc = half ? acc1 : acc0;
#pragma unroll
    for (int t = 0; t < 4; ++t) {
#pragma unroll
      for (int r = 0; r < 8; ++r) {
        int m  = mtile2 * 32 + half * 16 + r + 8 * hi;  // global row (b*S+s)
        int nf = ngrp * 64 + t * 16 + nlo;              // output feature
        float v = acc[t][r] + bias[nf];
        if (MODE == 0) {
          int b = m >> 11, s = m & (SDIM - 1), h = nf >> 6, d = nf & (DDIM - 1);
          out_bf[(((size_t)(b * HDIM + h) * SDIM + s) * DDIM) + d] = (bf16_t)v;
        } else if (MODE == 1) {
          int b = m >> 11, s = m & (SDIM - 1), h = nf >> 6, d = nf & (DDIM - 1);
          out_bf[(((size_t)(b * HDIM + h) * DDIM + d) * SDIM) + s] = (bf16_t)v;
        } else {
          out_f[(size_t)m * EDIM + nf] = v + resid[(size_t)m * EDIM + nf];
        }
      }
    }
  }
}

// ---------------------------------------------------------------------------
// 4) Causal flash attention. One wave owns one (b, h, 16-query tile).
//    Q,K: [B,H,S,D] bf16 ; Vt: [B,H,D,S] bf16 ; O: [B*S, H*D] bf16
//    Tile-wide running max (scalar) + row-sums computed by WMMA against a
//    ones matrix (sums land replicated in C-layout, matching the state).
// ---------------------------------------------------------------------------
__global__ void attn_kernel(const bf16_t* __restrict__ Q,
                            const bf16_t* __restrict__ K,
                            const bf16_t* __restrict__ Vt,
                            bf16_t* __restrict__ O) {
  __shared__ bf16_t pbuf[8][16 * 32];     // per-wave P staging (C->A transpose)
  int wid  = threadIdx.x >> 5;
  int w    = blockIdx.x * 8 + wid;
  int i    = w & 127;          // query tile
  int h    = (w >> 7) & 15;
  int b    = w >> 11;
  int lane = threadIdx.x & 31;
  int nlo  = lane & 15;
  int hi   = lane >> 4;

  const bf16_t* qb = Q + ((size_t)(b * HDIM + h) * SDIM + i * 16) * DDIM;
  v16bf qa0 = load_frag_a(qb, DDIM);          // K-dim 0..31
  v16bf qa1 = load_frag_a(qb + 32, DDIM);     // K-dim 32..63

  const bf16_t* kbase = K  + (size_t)(b * HDIM + h) * SDIM * DDIM;
  const bf16_t* vbase = Vt + (size_t)(b * HDIM + h) * DDIM * SDIM;
  bf16_t* pl = pbuf[wid];

  v16bf ones;
#pragma unroll
  for (int e = 0; e < 16; ++e) ones[e] = (bf16_t)1.0f;

  v8f oacc[4] = {};
  float lrow[8];
#pragma unroll
  for (int r = 0; r < 8; ++r) lrow[r] = 0.f;
  float mrun = -1e30f;

  int nblocks = (i >> 1) + 1;               // 32-key blocks (causal)
  for (int jb = 0; jb < nblocks; ++jb) {
    // ---- scores: two 16x16 tiles over the 32-key block ----
    v8f sacc[2];
#pragma unroll
    for (int t = 0; t < 2; ++t) {
      int kt = jb * 32 + t * 16;
      v8f s = {};
      v16bf kb0 = load_frag_b(kbase + (size_t)kt * DDIM, DDIM);
      s = wmma_bf16(qa0, kb0, s);
      v16bf kb1 = load_frag_b(kbase + (size_t)kt * DDIM + 32, DDIM);
      s = wmma_bf16(qa1, kb1, s);
#pragma unroll
      for (int r = 0; r < 8; ++r) {
        int kg = kt + nlo;
        int qg = i * 16 + r + 8 * hi;
        float sv = s[r] * 0.125f;            // 1/sqrt(64)
        sacc[t][r] = (kg > qg) ? -1e30f : sv;
      }
    }
    // ---- tile-wide max (VALU tree + one 32-lane butterfly) ----
    float mx = fmaxf(sacc[0][0], sacc[1][0]);
#pragma unroll
    for (int r = 1; r < 8; ++r)
      mx = fmaxf(mx, fmaxf(sacc[0][r], sacc[1][r]));
#pragma unroll
    for (int off = 16; off; off >>= 1)
      mx = fmaxf(mx, __shfl_xor(mx, off, 32));
    float mnew = fmaxf(mrun, mx);
    float corr = __expf(mrun - mnew);
    mrun = mnew;

    // ---- P = exp(s - m), staged to LDS for C->A layout transpose ----
#pragma unroll
    for (int t = 0; t < 2; ++t)
#pragma unroll
      for (int r = 0; r < 8; ++r) {
        float p = __expf(sacc[t][r] - mnew);
        pl[(r + 8 * hi) * 32 + t * 16 + nlo] = (bf16_t)p;
      }
    asm volatile("s_wait_dscnt 0x0" ::: "memory");
    v16bf pa = load_frag_a(pl, 32);

    // ---- row sums via WMMA against ones: lsum[r] replicated per lane ----
    v8f z = {};
    v8f lsum = wmma_bf16(pa, ones, z);
#pragma unroll
    for (int r = 0; r < 8; ++r) lrow[r] = lrow[r] * corr + lsum[r];

    // ---- rescale O, then O += P(16x32) * V(32x64) ----
#pragma unroll
    for (int t4 = 0; t4 < 4; ++t4)
#pragma unroll
      for (int r = 0; r < 8; ++r) oacc[t4][r] *= corr;
#pragma unroll
    for (int t4 = 0; t4 < 4; ++t4) {
      v16bf vb = load_frag_b(vbase + (size_t)(t4 * 16) * SDIM + jb * 32, SDIM);
      oacc[t4] = wmma_bf16(pa, vb, oacc[t4]);
    }
  }

  // ---- normalize + write [B*S, H*D] bf16 ----
#pragma unroll
  for (int t4 = 0; t4 < 4; ++t4) {
#pragma unroll
    for (int r = 0; r < 8; ++r) {
      int m  = b * SDIM + i * 16 + r + 8 * hi;
      int nf = h * DDIM + t4 * 16 + nlo;
      O[(size_t)m * (HDIM * DDIM) + nf] = (bf16_t)(oacc[t4][r] / lrow[r]);
    }
  }
}

// ---------------------------------------------------------------------------
// Launch
// ---------------------------------------------------------------------------
extern "C" void kernel_launch(void* const* d_in, const int* in_sizes, int n_in,
                              void* d_out, int out_size, void* d_ws, size_t ws_size,
                              hipStream_t stream) {
  const float* x    = (const float*)d_in[0];
  const float* ln_g = (const float*)d_in[1];
  const float* ln_b = (const float*)d_in[2];
  const float* wq   = (const float*)d_in[3];
  const float* bq   = (const float*)d_in[4];
  const float* wk   = (const float*)d_in[5];
  const float* bk   = (const float*)d_in[6];
  const float* wv   = (const float*)d_in[7];
  const float* bv   = (const float*)d_in[8];
  const float* wo   = (const float*)d_in[9];
  const float* bo   = (const float*)d_in[10];
  float* out = (float*)d_out;

  uint8_t* ws = (uint8_t*)d_ws;
  const size_t SZ_ACT = (size_t)ROWS * EDIM * sizeof(bf16_t);  // 8 MB
  const size_t SZ_W   = (size_t)EDIM * EDIM * sizeof(bf16_t);  // 2 MB
  bf16_t* xn   = (bf16_t*)(ws);
  bf16_t* wqt  = (bf16_t*)(ws + SZ_ACT);
  bf16_t* wkt  = (bf16_t*)(ws + SZ_ACT + SZ_W);
  bf16_t* wvt  = (bf16_t*)(ws + SZ_ACT + 2 * SZ_W);
  bf16_t* wot  = (bf16_t*)(ws + SZ_ACT + 3 * SZ_W);
  bf16_t* qb   = (bf16_t*)(ws + SZ_ACT + 4 * SZ_W);
  bf16_t* kb   = (bf16_t*)(ws + 2 * SZ_ACT + 4 * SZ_W);
  bf16_t* vt   = (bf16_t*)(ws + 3 * SZ_ACT + 4 * SZ_W);
  bf16_t* attn = (bf16_t*)(ws + 4 * SZ_ACT + 4 * SZ_W);

  // 1) LayerNorm -> bf16 activations
  ln_kernel<<<ROWS, 256, 0, stream>>>(x, ln_g, ln_b, xn);

  // 2) transpose+convert the four weights
  wtrans_kernel<<<dim3(32, 32, 4), 256, 0, stream>>>(wq, wk, wv, wo,
                                                     wqt, wkt, wvt, wot);

  // 3) Q, K, V projections (WMMA)
  gemm_kernel<0><<<256, 256, 0, stream>>>(xn, wqt, bq, nullptr, qb, nullptr);
  gemm_kernel<0><<<256, 256, 0, stream>>>(xn, wkt, bk, nullptr, kb, nullptr);
  gemm_kernel<1><<<256, 256, 0, stream>>>(xn, wvt, bv, nullptr, vt, nullptr);

  // 4) causal flash attention (WMMA)
  attn_kernel<<<512, 256, 0, stream>>>(qb, kb, vt, attn);

  // 5) output projection + bias + residual -> f32 d_out (WMMA)
  gemm_kernel<2><<<256, 256, 0, stream>>>(attn, wot, bo, x, nullptr, out);
}
